// AveragePrecision_9491877724869
// MI455X (gfx1250) — compile-verified, matrix operand order checked by reading, not run on hardware
//
#include <hip/hip_runtime.h>
#include <hip/hip_bf16.h>

// ---------------------------------------------------------------------------
// AveragePrecision on MI455X (gfx1250)
//   Phase 1: per-WG LDS-privatized 64K-bin histogram (u16 pairs packed in u32,
//            ds_add_u32 atomics), int4 (b128) streaming loads, b128 flush.
//   Phase 2: atomic-free reduction of 128 partials -> inter[C][C] as f32.
//   Phase 3: one WG; cnt_gt/cnt_pr sums; division-free IoU thresholds
//            (2*inter vs union, exact integer-in-f32 compares) -> 0/1 f16
//            indicator tiles row-summed with v_wmma_f32_16x16x32_f16 against
//            an all-ones B (exact); mean precision via LDS f32 atomics.
// ---------------------------------------------------------------------------

#define C_LAB   256
#define NBINS   (C_LAB * C_LAB)   // 65536 bins
#define NWORDS  (NBINS / 2)       // 32768 packed u32 words (2 x u16 bins)
#define HIST_WGS 128
#define HIST_THR 512
#define IOU_TH  0.5f

typedef __attribute__((ext_vector_type(16))) _Float16 v16h;
typedef __attribute__((ext_vector_type(8)))  float    v8f;

// ---------------- Phase 1: per-workgroup privatized histogram ---------------
__global__ void __launch_bounds__(HIST_THR)
ap_hist_partial(const int* __restrict__ input, const int* __restrict__ target,
                unsigned* __restrict__ partial, int n)
{
    extern __shared__ unsigned lds_hist[];   // NWORDS u32 = 128 KB
    const int t  = threadIdx.x;
    const int wg = blockIdx.x;

#pragma unroll
    for (int j = 0; j < NWORDS / HIST_THR; ++j)
        lds_hist[t + j * HIST_THR] = 0u;
    __syncthreads();

    const int perWG = n / HIST_WGS;              // 65536 points per WG
    const int base  = wg * perWG;
    const int iters = perWG / (HIST_THR * 4);    // 32 int4 iterations/thread
    const int4* in4 = (const int4*)(input  + base);
    const int4* tg4 = (const int4*)(target + base);

    for (int i = 0; i < iters; ++i) {
        int4 a = in4[i * HIST_THR + t];          // global_load_b128
        int4 b = tg4[i * HIST_THR + t];
        int i0 = b.x * C_LAB + a.x;
        int i1 = b.y * C_LAB + a.y;
        int i2 = b.z * C_LAB + a.z;
        int i3 = b.w * C_LAB + a.w;
        atomicAdd(&lds_hist[i0 >> 1], (i0 & 1) ? 0x10000u : 1u);  // ds_add_u32
        atomicAdd(&lds_hist[i1 >> 1], (i1 & 1) ? 0x10000u : 1u);
        atomicAdd(&lds_hist[i2 >> 1], (i2 & 1) ? 0x10000u : 1u);
        atomicAdd(&lds_hist[i3 >> 1], (i3 & 1) ? 0x10000u : 1u);
    }
    __syncthreads();

    // atomic-free flush of the full private histogram (plain b128 stores)
    uint4*       dst = (uint4*)(partial + (size_t)wg * NWORDS);
    const uint4* src = (const uint4*)lds_hist;
#pragma unroll
    for (int j = 0; j < (NWORDS / 4) / HIST_THR; ++j)
        dst[t + j * HIST_THR] = src[t + j * HIST_THR];
}

// ---------------- Phase 2: reduce partials -> inter (f32, exact) ------------
__global__ void __launch_bounds__(256)
ap_reduce(const unsigned* __restrict__ partial, float* __restrict__ inter)
{
    const int w = blockIdx.x * 256 + threadIdx.x;   // 0 .. NWORDS-1
    unsigned lo = 0u, hi = 0u;
    for (int g = 0; g < HIST_WGS; ++g) {
        unsigned v = partial[(size_t)g * NWORDS + w];
        lo += v & 0xFFFFu;
        hi += v >> 16;
    }
    inter[2 * w]     = (float)lo;   // counts < 2^24 -> exact in f32
    inter[2 * w + 1] = (float)hi;
}

// ---------------- Phase 3: IoU -> TP/FP via WMMA -> mean precision ----------
__global__ void __launch_bounds__(256)
ap_final(const float* __restrict__ inter, float* __restrict__ out)
{
    __shared__ float s_cg[C_LAB], s_cp[C_LAB], s_tp[C_LAB], s_fp[C_LAB];
    __shared__ float s_acc[2];
    const int t = threadIdx.x;

    // row / column sums of the confusion matrix
    float cg = 0.f, cp = 0.f;
    for (int j = 0; j < C_LAB; ++j) {
        cg += inter[t * C_LAB + j];
        cp += inter[j * C_LAB + t];
    }
    s_cg[t] = cg;
    s_cp[t] = cp;
    if (t == 0) { s_acc[0] = 0.f; s_acc[1] = 0.f; }
    __syncthreads();

    const int lane = t & 31;
    const int wave = t >> 5;
    const int mrow = lane & 15;           // A-matrix row M for this lane
    const int kHi  = (lane >> 4) * 8;     // K offset for lanes 16..31

    const _Float16 one  = (_Float16)1.0f;
    const _Float16 zero = (_Float16)0.0f;

    v16h bOnes;
#pragma unroll
    for (int h = 0; h < 16; ++h) bOnes[h] = one;

    // 16 row-groups of 16 gt rows; each of the 8 waves handles two
#pragma unroll
    for (int rr = 0; rr < 2; ++rr) {
        const int rowbase = (wave * 2 + rr) * 16;
        const int g   = rowbase + mrow;
        const float cgv = s_cg[g];

        v8f cTP = {};
        v8f cFP = {};
        for (int k0 = 0; k0 < C_LAB; k0 += 32) {
            v16h aTP, aFP;
#pragma unroll
            for (int h = 0; h < 16; ++h) {
                // ISA 16-bit A layout: lane<16 holds K in {0..7,16..23},
                //                      lane>=16 holds K in {8..15,24..31}
                const int k = (h < 8 ? h : h + 8) + kHi;
                const int p = k0 + k;
                const float it = inter[g * C_LAB + p];
                const float un = cgv + s_cp[p] - it;   // exact integer in f32
                const float it2 = it + it;
                // Division-free, bit-exact vs reference:
                //   iou >= 0.5  <=>  un > 0 && 2*it >= un
                //   0 < iou < .5 <=> it > 0 && 2*it <  un
                aTP[h] = (un > 0.f && it2 >= un) ? one : zero;
                aFP[h] = (it > 0.f && it2 <  un) ? one : zero;
            }
            // row-sum of 0/1 indicators: exact f16 x ones -> f32 accum
            cTP = __builtin_amdgcn_wmma_f32_16x16x32_f16(
                false, aTP, false, bOnes, (short)0, cTP, false, false);
            cFP = __builtin_amdgcn_wmma_f32_16x16x32_f16(
                false, aFP, false, bOnes, (short)0, cFP, false, false);
        }
        // D layout: VGPR r -> (M=r, N=lane) for lanes 0..15, (M=8+r, N=lane-16)
        // for lanes 16..31; take column N=0 (lanes 0 and 16).
        if (lane == 0) {
#pragma unroll
            for (int r = 0; r < 8; ++r) {
                s_tp[rowbase + r] = cTP[r];
                s_fp[rowbase + r] = cFP[r];
            }
        }
        if (lane == 16) {
#pragma unroll
            for (int r = 0; r < 8; ++r) {
                s_tp[rowbase + 8 + r] = cTP[r];
                s_fp[rowbase + 8 + r] = cFP[r];
            }
        }
    }
    __syncthreads();

    const float tp = s_tp[t];
    const float fp = s_fp[t];
    const bool  present = s_cg[t] > 0.f;
    const float prec = present ? tp / fmaxf(tp + fp, 1.f) : 0.f;
    atomicAdd(&s_acc[0], prec);                       // ds_add_f32
    atomicAdd(&s_acc[1], present ? 1.f : 0.f);
    __syncthreads();
    if (t == 0) out[0] = s_acc[0] / fmaxf(s_acc[1], 1.f);
}

// ---------------------------------------------------------------------------
extern "C" void kernel_launch(void* const* d_in, const int* in_sizes, int n_in,
                              void* d_out, int out_size, void* d_ws, size_t ws_size,
                              hipStream_t stream)
{
    const int* input  = (const int*)d_in[0];
    const int* target = (const int*)d_in[1];
    const int  n      = in_sizes[0];     // 8,388,608

    // workspace layout: [HIST_WGS * NWORDS u32 partials | NBINS f32 inter]
    unsigned* partial = (unsigned*)d_ws;
    float*    inter   = (float*)((char*)d_ws + (size_t)HIST_WGS * NWORDS * sizeof(unsigned));

    ap_hist_partial<<<HIST_WGS, HIST_THR, NWORDS * sizeof(unsigned), stream>>>(
        input, target, partial, n);
    ap_reduce<<<HIST_WGS, 256, 0, stream>>>(partial, inter);
    ap_final<<<1, 256, 0, stream>>>(inter, (float*)d_out);
}
